// ExampleNet_24094766530645
// MI455X (gfx1250) — matrix operand order repaired and checked by reading, not compiled, not run on hardware
//
#include <hip/hip_runtime.h>

typedef __attribute__((ext_vector_type(8)))  float  v8f;
typedef __attribute__((ext_vector_type(8)))  __bf16 v8bf;
typedef __attribute__((ext_vector_type(16))) __bf16 v16bf;

#define NNODES 50000
#define NEDGES 400000
#define NGRAPHS 64

// ---- helpers -------------------------------------------------------------

__device__ __forceinline__ v16bf lds_load16(const __bf16* p) {
  // 16 contiguous bf16 (32B) from LDS -> two ds_load_b128
  v8bf lo = *(const v8bf*)p;
  v8bf hi = *(const v8bf*)(p + 8);
  return __builtin_shufflevector(lo, hi, 0,1,2,3,4,5,6,7,8,9,10,11,12,13,14,15);
}

__device__ __forceinline__ v8f wmma_bf16(v16bf a, v16bf b, v8f c) {
  // D = A(16x32 bf16) * B(32x16 bf16) + C(16x16 f32)
  return __builtin_amdgcn_wmma_f32_16x16x32_bf16(false, a, false, b, (short)0, c,
                                                 false, false);
}

__global__ void zero_f32(float* __restrict__ p, int n) {
  int i = blockIdx.x * blockDim.x + threadIdx.x;
  if (i < n) p[i] = 0.0f;
}

// ---- conv1 edge kernel: in=16 feats, hidden=32, out=32 -------------------
// One wave handles a tile of 16 edges. msg(16x32) = Z(16x512)@W2r(512x32)+x@B2r
// Z[e, i*32+k] = x[src[e], i] * relu(eattr@w1+b1)[e,k]
__global__ __launch_bounds__(256) void conv1_edge(
    const float* __restrict__ x,     // NNODES x 16
    const int*   __restrict__ src,
    const int*   __restrict__ dst,
    const float* __restrict__ ea_g,  // NEDGES x 8
    const float* __restrict__ w1,    // 8 x 32
    const float* __restrict__ b1v,   // 32
    const float* __restrict__ w2,    // 32 x 512  (k, i*32+o)
    const float* __restrict__ b2v,   // 512       (i*32+o)
    float*       __restrict__ agg)   // NNODES x 32 (pre-zeroed, atomics)
{
  __shared__ __align__(16) __bf16 w2t[512 * 32];  // [col = i*32+o][k], bf16
  __builtin_prefetch(w2, 0, 0);
  for (int idx = threadIdx.x; idx < 512 * 32; idx += 256) {
    int col = idx >> 5, k = idx & 31;
    w2t[idx] = (__bf16)w2[k * 512 + col];
  }
  __syncthreads();

  const int lane = threadIdx.x & 31;
  const int wave = threadIdx.x >> 5;
  const int lh   = lane >> 4;     // lane half (ISA A/B layout)
  const int n    = lane & 15;     // N column; also A-row (edge-in-tile) m
  const int tile = (blockIdx.x * 8 + wave) * 16;
  if (tile >= NEDGES) return;     // wave-uniform: EXEC stays all-ones for WMMA

  const int e = tile + n;
  const int s = src[e];

  __bf16 xbf[16];
#pragma unroll
  for (int i = 0; i < 16; ++i) xbf[i] = (__bf16)x[s * 16 + i];

  float ea[8];
#pragma unroll
  for (int f = 0; f < 8; ++f) ea[f] = ea_g[e * 8 + f];

  // this lane's 16 hidden values, already in A-fragment element order:
  // element j <-> k = j + 16*lh ; pre-converted to bf16
  v16bf hvbf;
#pragma unroll
  for (int j = 0; j < 16; ++j) {
    int k = j + 16 * lh;
    float acc = b1v[k];
#pragma unroll
    for (int f = 0; f < 8; ++f) acc += ea[f] * w1[f * 32 + k];
    hvbf[j] = (__bf16)fmaxf(acc, 0.0f);
  }

  v8f c0 = {};  // output cols 0..15
  v8f c1 = {};  // output cols 16..31
#pragma unroll
  for (int c = 0; c < 16; ++c) {  // K-chunk c covers Z cols [32c,32c+31] => i==c
    // A fragment: Z row slice = x[e,c] * h[e,:]  (8x v_pk_mul_bf16)
    v16bf a = hvbf * xbf[c];
    // B element j <-> K = j + 16*lh, col = c*32 + oh*16 + n  (contiguous in k)
    v16bf bl = lds_load16(&w2t[(c * 32 + n) * 32 + 16 * lh]);
    v16bf bh = lds_load16(&w2t[(c * 32 + 16 + n) * 32 + 16 * lh]);
    c0 = wmma_bf16(a, bl, c0);
    c1 = wmma_bf16(a, bh, c1);
  }

  // bias: msg += x(16x16, K zero-padded to 32) @ reshape(b2,(16,32))
  {
    v16bf a = {};
#pragma unroll
    for (int j = 0; j < 16; ++j) if (lh == 0) a[j] = xbf[j];  // K<16 valid
    v16bf bl = {}; v16bf bh = {};
    if (lh == 0) {  // K = j+16*lh valid only for lh==0
#pragma unroll
      for (int j = 0; j < 16; ++j) {
        bl[j] = (__bf16)b2v[j * 32 + n];
        bh[j] = (__bf16)b2v[j * 32 + 16 + n];
      }
    }
    c0 = wmma_bf16(a, bl, c0);
    c1 = wmma_bf16(a, bh, c1);
  }

  // scatter: C element r <-> row M = r + 8*lh, col N = n (+16 for c1)
#pragma unroll
  for (int r = 0; r < 8; ++r) {
    int d = dst[tile + r + 8 * lh];
    atomicAdd(&agg[d * 32 + n],      c0[r]);
    atomicAdd(&agg[d * 32 + 16 + n], c1[r]);
  }
}

// ---- conv2 edge kernel: in=32 feats, hidden=32, out=16 -------------------
__global__ __launch_bounds__(256) void conv2_edge(
    const float* __restrict__ xin,   // NNODES x 32 (out1)
    const int*   __restrict__ src,
    const int*   __restrict__ dst,
    const float* __restrict__ ea_g,  // NEDGES x 8
    const float* __restrict__ w1,    // 8 x 32
    const float* __restrict__ b1v,   // 32
    const float* __restrict__ w2,    // 32 x 512  (k, i*16+o)
    const float* __restrict__ b2v,   // 512       (i*16+o)
    float*       __restrict__ agg)   // NNODES x 16 (pre-zeroed, atomics)
{
  __shared__ __align__(16) __bf16 w2t[512 * 32];  // [col = i*16+o][k]
  __builtin_prefetch(w2, 0, 0);
  for (int idx = threadIdx.x; idx < 512 * 32; idx += 256) {
    int col = idx >> 5, k = idx & 31;
    w2t[idx] = (__bf16)w2[k * 512 + col];
  }
  __syncthreads();

  const int lane = threadIdx.x & 31;
  const int wave = threadIdx.x >> 5;
  const int lh   = lane >> 4;
  const int n    = lane & 15;
  const int tile = (blockIdx.x * 8 + wave) * 16;
  if (tile >= NEDGES) return;

  const int e = tile + n;
  const int s = src[e];

  __bf16 xbf[32];
#pragma unroll
  for (int i = 0; i < 32; ++i) xbf[i] = (__bf16)xin[s * 32 + i];

  float ea[8];
#pragma unroll
  for (int f = 0; f < 8; ++f) ea[f] = ea_g[e * 8 + f];

  v16bf hvbf;
#pragma unroll
  for (int j = 0; j < 16; ++j) {
    int k = j + 16 * lh;
    float acc = b1v[k];
#pragma unroll
    for (int f = 0; f < 8; ++f) acc += ea[f] * w1[f * 32 + k];
    hvbf[j] = (__bf16)fmaxf(acc, 0.0f);
  }

  v8f c0 = {};  // output cols 0..15 (out=16 exactly one N-tile)
#pragma unroll
  for (int c = 0; c < 32; ++c) {  // 32 K-chunks, i == c per chunk
    v16bf a = hvbf * xbf[c];
    v16bf b = lds_load16(&w2t[(c * 16 + n) * 32 + 16 * lh]);
    c0 = wmma_bf16(a, b, c0);
  }

  // bias: msg += x(16x32) @ reshape(b2,(32,16)) — exactly one K=32 WMMA
  {
    v16bf a;
#pragma unroll
    for (int j = 0; j < 16; ++j) {
      int k = j + 16 * lh;
      a[j] = xbf[k];
    }
    v16bf b;
#pragma unroll
    for (int j = 0; j < 16; ++j) {
      int k = j + 16 * lh;
      b[j] = (__bf16)b2v[k * 16 + n];
    }
    c0 = wmma_bf16(a, b, c0);
  }

#pragma unroll
  for (int r = 0; r < 8; ++r) {
    int d = dst[tile + r + 8 * lh];
    atomicAdd(&agg[d * 16 + n], c0[r]);
  }
}

// ---- node update 1: out1 = relu(agg1 + x@root + bias), 50000x32 ----------
__global__ void node1_update(const float* __restrict__ x,
                             const float* __restrict__ agg,
                             const float* __restrict__ root,  // 16x32
                             const float* __restrict__ bias,  // 32
                             float* __restrict__ out1) {
  int idx = blockIdx.x * blockDim.x + threadIdx.x;
  if (idx >= NNODES * 32) return;
  int v = idx >> 5, k = idx & 31;
  float acc = agg[idx] + bias[k];
#pragma unroll
  for (int i = 0; i < 16; ++i) acc += x[v * 16 + i] * root[i * 32 + k];
  out1[idx] = fmaxf(acc, 0.0f);
}

// ---- node update 2 + global_add_pool: 50000x16 ---------------------------
__global__ void node2_update_pool(const float* __restrict__ h1,   // 50000x32
                                  const float* __restrict__ agg,  // 50000x16
                                  const float* __restrict__ root, // 32x16
                                  const float* __restrict__ bias, // 16
                                  const int*   __restrict__ batch,
                                  float* __restrict__ pooled) {   // 64x16
  int idx = blockIdx.x * blockDim.x + threadIdx.x;
  if (idx >= NNODES * 16) return;
  int v = idx >> 4, k = idx & 15;
  float acc = agg[idx] + bias[k];
#pragma unroll
  for (int i = 0; i < 32; ++i) acc += h1[v * 32 + i] * root[i * 16 + k];
  acc = fmaxf(acc, 0.0f);
  atomicAdd(&pooled[batch[v] * 16 + k], acc);
}

// ---- head: 64 graphs, 16 -> 32 relu -> 1 ---------------------------------
__global__ void head_mlp(const float* __restrict__ pooled,  // 64x16
                         const float* __restrict__ fc1_w,   // 16x32
                         const float* __restrict__ fc1_b,   // 32
                         const float* __restrict__ out_w,   // 32x1
                         const float* __restrict__ out_b,   // 1
                         float* __restrict__ out) {         // 64
  int g = threadIdx.x;
  if (g >= NGRAPHS) return;
  float pg[16];
#pragma unroll
  for (int i = 0; i < 16; ++i) pg[i] = pooled[g * 16 + i];
  float res = out_b[0];
#pragma unroll
  for (int k = 0; k < 32; ++k) {
    float t = fc1_b[k];
#pragma unroll
    for (int i = 0; i < 16; ++i) t += pg[i] * fc1_w[i * 32 + k];
    res += fmaxf(t, 0.0f) * out_w[k];
  }
  out[g] = res;
}

// ---- launch --------------------------------------------------------------
extern "C" void kernel_launch(void* const* d_in, const int* in_sizes, int n_in,
                              void* d_out, int out_size, void* d_ws, size_t ws_size,
                              hipStream_t stream) {
  const float* x       = (const float*)d_in[0];
  const int*   eidx    = (const int*)  d_in[1];
  const float* eattr   = (const float*)d_in[2];
  const int*   batch   = (const int*)  d_in[3];
  const float* c1_w1   = (const float*)d_in[4];
  const float* c1_b1   = (const float*)d_in[5];
  const float* c1_w2   = (const float*)d_in[6];
  const float* c1_b2   = (const float*)d_in[7];
  const float* c1_root = (const float*)d_in[8];
  const float* c1_bias = (const float*)d_in[9];
  const float* c2_w1   = (const float*)d_in[10];
  const float* c2_b1   = (const float*)d_in[11];
  const float* c2_w2   = (const float*)d_in[12];
  const float* c2_b2   = (const float*)d_in[13];
  const float* c2_root = (const float*)d_in[14];
  const float* c2_bias = (const float*)d_in[15];
  const float* fc1_w   = (const float*)d_in[16];
  const float* fc1_b   = (const float*)d_in[17];
  const float* out_w   = (const float*)d_in[18];
  const float* out_b   = (const float*)d_in[19];

  const int* src = eidx;
  const int* dst = eidx + NEDGES;

  float* agg1   = (float*)d_ws;            // NNODES*32
  float* out1   = agg1 + NNODES * 32;      // NNODES*32
  float* agg2   = agg1;                    // NNODES*16 (reuses agg1 region)
  float* pooled = out1 + NNODES * 32;      // 64*16
  float* outp   = (float*)d_out;

  const int edgeBlocks = NEDGES / (16 * 8);  // 3125, exact

  zero_f32<<<(NNODES * 32 + 255) / 256, 256, 0, stream>>>(agg1, NNODES * 32);
  conv1_edge<<<edgeBlocks, 256, 0, stream>>>(x, src, dst, eattr,
                                             c1_w1, c1_b1, c1_w2, c1_b2, agg1);
  node1_update<<<(NNODES * 32 + 255) / 256, 256, 0, stream>>>(x, agg1, c1_root,
                                                              c1_bias, out1);
  zero_f32<<<(NNODES * 16 + 255) / 256, 256, 0, stream>>>(agg2, NNODES * 16);
  conv2_edge<<<edgeBlocks, 256, 0, stream>>>(out1, src, dst, eattr,
                                             c2_w1, c2_b1, c2_w2, c2_b2, agg2);
  zero_f32<<<(NGRAPHS * 16 + 255) / 256, 256, 0, stream>>>(pooled, NGRAPHS * 16);
  node2_update_pool<<<(NNODES * 16 + 255) / 256, 256, 0, stream>>>(
      out1, agg2, c2_root, c2_bias, batch, pooled);
  head_mlp<<<1, 64, 0, stream>>>(pooled, fc1_w, fc1_b, out_w, out_b, outp);
}
